// LPDNet_25761213841840
// MI455X (gfx1250) — matrix operand order repaired and checked by reading, not compiled, not run on hardware
//
#include <hip/hip_runtime.h>

#define BATCH 4
#define NPTS  4096
#define KNBR  20
#define EPSF  1e-5f
#define SLOPE 0.01f
#define NT    8
#define COLS  (NT*KNBR)   // 160
#define CSTR  (COLS+8)    // 168 (LDS pad)

typedef _Float16 half_t;
typedef _Float16 v16h __attribute__((ext_vector_type(16)));
typedef float    v8f  __attribute__((ext_vector_type(8)));

__device__ __forceinline__ float leakyf(float v){ return v >= 0.f ? v : SLOPE * v; }

// ---------------- [B,1,N,3] -> [B,3,N]
__global__ void k_transpose(const float* __restrict__ x, float* __restrict__ pts)
{
  int i = blockIdx.x * blockDim.x + threadIdx.x;
  if (i >= BATCH * NPTS) return;
  int b = i / NPTS, n = i % NPTS;
  #pragma unroll
  for (int c = 0; c < 3; ++c)
    pts[((size_t)b*3 + c)*NPTS + n] = x[((size_t)b*NPTS + n)*3 + c];
}

// ---------------- 3-channel 1x1 conv (tiny; WMMA would be 90% zero padding)
__global__ void k_conv_small(const float* __restrict__ W,     // [O,3]
                             const float* __restrict__ bias,
                             const float* __restrict__ bnG, const float* __restrict__ bnB,
                             const float* __restrict__ bnM, const float* __restrict__ bnV,
                             int act,
                             const float* __restrict__ in,    // [B,3,N]
                             float* __restrict__ out, int O)
{
  int i = blockIdx.x * blockDim.x + threadIdx.x;
  if (i >= BATCH * O * NPTS) return;
  int n = i % NPTS, o = (i / NPTS) % O, b = i / (NPTS * O);
  const float* inb = in + (size_t)b*3*NPTS;
  float s = bias ? bias[o] : 0.f;
  #pragma unroll
  for (int c = 0; c < 3; ++c) s += W[o*3 + c] * inb[(size_t)c*NPTS + n];
  s = (s - bnM[o]) * (bnG[o] * rsqrtf(bnV[o] + EPSF)) + bnB[o];
  s = (act == 1) ? fmaxf(s, 0.f) : leakyf(s);
  out[(size_t)b*O*NPTS + (size_t)o*NPTS + n] = s;
}

// ---------------- generic fused WMMA GEMM (C must be a multiple of 32):
// out[b,o,n] = act(bn(bias + sum_c W[o, w_off+c] * in(c)[b,:,n]))
// input channel c maps: c<c1 -> in0[c], c<c2 -> in1[c-c1], else in2[c-c2]
// c1,c2 must be multiples of 16. One wave computes a 32x32 output tile.
__global__ void k_gemm_wmma(
    const float* __restrict__ W, int w_ld, int w_off,
    const float* __restrict__ in0, size_t s0,
    const float* __restrict__ in1, size_t s1,
    const float* __restrict__ in2, size_t s2,
    int c1, int c2,
    const float* __restrict__ bias,
    const float* __restrict__ bnG, const float* __restrict__ bnB,
    const float* __restrict__ bnM, const float* __restrict__ bnV,
    int act,
    float* __restrict__ out, size_t outbs,
    int O, int C, int N)
{
  const int n0 = blockIdx.x * 32;
  const int o0 = blockIdx.y * 32;
  const int b  = blockIdx.z;
  const int lane = threadIdx.x;      // wave32
  const int g  = lane >> 4;
  const int lm = lane & 15;

  v8f acc00 = {}, acc01 = {}, acc10 = {}, acc11 = {};
  for (int k0 = 0; k0 < C; k0 += 32) {
    // ---- A fragments (weights), two o-tiles; per lane: run [k0+16g, +16)
    v16h a0, a1;
    #pragma unroll
    for (int t = 0; t < 2; ++t) {
      const float* wr = W + (size_t)(o0 + t*16 + lm)*w_ld + w_off + k0 + 16*g;
      float4 q0 = *(const float4*)(wr);
      float4 q1 = *(const float4*)(wr + 4);
      float4 q2 = *(const float4*)(wr + 8);
      float4 q3 = *(const float4*)(wr + 12);
      v16h a;
      a[0]=(half_t)q0.x;  a[1]=(half_t)q0.y;  a[2]=(half_t)q0.z;  a[3]=(half_t)q0.w;
      a[4]=(half_t)q1.x;  a[5]=(half_t)q1.y;  a[6]=(half_t)q1.z;  a[7]=(half_t)q1.w;
      a[8]=(half_t)q2.x;  a[9]=(half_t)q2.y;  a[10]=(half_t)q2.z; a[11]=(half_t)q2.w;
      a[12]=(half_t)q3.x; a[13]=(half_t)q3.y; a[14]=(half_t)q3.z; a[15]=(half_t)q3.w;
      if (t == 0) a0 = a; else a1 = a;
    }
    // ---- B fragments, two n-tiles; 16 consecutive channels starting at k0+16g
    const int cbase = k0 + 16*g;
    const float* src; size_t bs; int cc;
    if (cbase < c1)      { src = in0; bs = s0; cc = cbase; }
    else if (cbase < c2) { src = in1; bs = s1; cc = cbase - c1; }
    else                 { src = in2; bs = s2; cc = cbase - c2; }
    const float* sp = src + bs*(size_t)b + (size_t)cc*N + n0 + lm;
    v16h b0, b1;
    #pragma unroll
    for (int h = 0; h < 16; ++h) {
      b0[h] = (half_t)sp[(size_t)h*N];
      b1[h] = (half_t)sp[(size_t)h*N + 16];
    }
    acc00 = __builtin_amdgcn_wmma_f32_16x16x32_f16(false, a0, false, b0, (short)0, acc00, false, false);
    acc01 = __builtin_amdgcn_wmma_f32_16x16x32_f16(false, a0, false, b1, (short)0, acc01, false, false);
    acc10 = __builtin_amdgcn_wmma_f32_16x16x32_f16(false, a1, false, b0, (short)0, acc10, false, false);
    acc11 = __builtin_amdgcn_wmma_f32_16x16x32_f16(false, a1, false, b1, (short)0, acc11, false, false);
  }
  // C/D layout: lane lm = col, VGPR r -> row r + 8*g
  #pragma unroll
  for (int to = 0; to < 2; ++to) {
    #pragma unroll
    for (int tn = 0; tn < 2; ++tn) {
      const v8f& acc = to ? (tn ? acc11 : acc10) : (tn ? acc01 : acc00);
      #pragma unroll
      for (int r = 0; r < 8; ++r) {
        const int o = o0 + to*16 + r + 8*g;
        const int n = n0 + tn*16 + lm;
        float v = acc[r];
        if (bias) v += bias[o];
        if (bnG)  v = (v - bnM[o]) * (bnG[o] * rsqrtf(bnV[o] + EPSF)) + bnB[o];
        if (act == 1) v = fmaxf(v, 0.f);
        else if (act == 2) v = leakyf(v);
        out[outbs*(size_t)b + (size_t)o*N + n] = v;
      }
    }
  }
}

// ---------------- TNet conv3 (128->1024) fused with relu(bn) and max over N
__global__ void k_conv_max(const float* __restrict__ W, const float* __restrict__ bias,
                           const float* __restrict__ bnG, const float* __restrict__ bnB,
                           const float* __restrict__ bnM, const float* __restrict__ bnV,
                           const float* __restrict__ in, float* __restrict__ gmax,
                           int O, int C, int N)
{
  int o = blockIdx.x, b = blockIdx.y;
  extern __shared__ float cm_sm[];
  float* wrow = cm_sm;          // C
  float* red  = cm_sm + C;      // blockDim
  for (int c = threadIdx.x; c < C; c += blockDim.x) wrow[c] = W[(size_t)o*C + c];
  __syncthreads();
  const float* inb = in + (size_t)b*C*N;
  const float bi = bias ? bias[o] : 0.f;
  const float sc = bnG[o] * rsqrtf(bnV[o] + EPSF);
  const float sh = bnB[o] - bnM[o]*sc;
  float best = -3.4e38f;
  for (int n = threadIdx.x; n < N; n += blockDim.x) {
    float s = bi;
    for (int c = 0; c < C; ++c) s += wrow[c] * inb[(size_t)c*N + n];
    s = fmaxf(s*sc + sh, 0.f);
    best = fmaxf(best, s);
  }
  red[threadIdx.x] = best;
  __syncthreads();
  for (int st = blockDim.x >> 1; st > 0; st >>= 1) {
    if (threadIdx.x < st) red[threadIdx.x] = fmaxf(red[threadIdx.x], red[threadIdx.x+st]);
    __syncthreads();
  }
  if (threadIdx.x == 0) gmax[(size_t)b*O + o] = red[0];
}

// ---------------- tiny FC: out[b,o] = act(bn(in[b,:]@W[o,:]+bias)) (+eye for fc3)
__global__ void k_fc(const float* __restrict__ W, const float* __restrict__ bias,
                     const float* __restrict__ bnG, const float* __restrict__ bnB,
                     const float* __restrict__ bnM, const float* __restrict__ bnV,
                     int act, int add_eye,
                     const float* __restrict__ in, float* __restrict__ out, int O, int C)
{
  int i = blockIdx.x * blockDim.x + threadIdx.x;
  if (i >= BATCH * O) return;
  int b = i / O, o = i % O;
  const float* inb = in + (size_t)b*C;
  float s = bias[o];
  for (int c = 0; c < C; ++c) s += W[(size_t)o*C + c] * inb[c];
  if (bnG) s = (s - bnM[o]) * (bnG[o] * rsqrtf(bnV[o] + EPSF)) + bnB[o];
  if (act == 1) s = fmaxf(s, 0.f);
  if (add_eye && (o/3) == (o%3)) s += 1.f;
  out[(size_t)b*O + o] = s;
}

// ---------------- h0[b,d,n] = sum_c pts[b,c,n]*trans[b,c,d]
__global__ void k_apply_trans(const float* __restrict__ pts, const float* __restrict__ trans,
                              float* __restrict__ h0)
{
  int i = blockIdx.x * blockDim.x + threadIdx.x;
  if (i >= BATCH * NPTS) return;
  int b = i / NPTS, n = i % NPTS;
  const float* tb = trans + (size_t)b*9;
  float p0 = pts[((size_t)b*3+0)*NPTS+n];
  float p1 = pts[((size_t)b*3+1)*NPTS+n];
  float p2 = pts[((size_t)b*3+2)*NPTS+n];
  #pragma unroll
  for (int d = 0; d < 3; ++d)
    h0[((size_t)b*3+d)*NPTS+n] = p0*tb[0*3+d] + p1*tb[1*3+d] + p2*tb[2*3+d];
}

// ---------------- exact fp32 kNN: per-row distances + repeated argmax (tie -> lowest idx)
__global__ void k_knn(const float* __restrict__ x, int C, int* __restrict__ idx)
{
  int n = blockIdx.x, b = blockIdx.y;
  extern __shared__ float knn_sm[];
  float* dist = knn_sm;                 // NPTS
  float* xn   = knn_sm + NPTS;          // 64
  float* rval = xn + 64;                // blockDim
  int*   ridx = (int*)(rval + blockDim.x);
  const float* xb = x + (size_t)b*C*NPTS;
  for (int c = threadIdx.x; c < C; c += blockDim.x) xn[c] = xb[(size_t)c*NPTS + n];
  __syncthreads();
  float xxn = 0.f;
  for (int c = 0; c < C; ++c) xxn += xn[c]*xn[c];
  for (int m = threadIdx.x; m < NPTS; m += blockDim.x) {
    float dot = 0.f, xxm = 0.f;
    for (int c = 0; c < C; ++c) { float v = xb[(size_t)c*NPTS + m]; dot += xn[c]*v; xxm += v*v; }
    dist[m] = 2.f*dot - xxn - xxm;      // neg squared distance
  }
  __syncthreads();
  for (int kk = 0; kk < KNBR; ++kk) {
    float bv = -3.4e38f; int bi = 0x7fffffff;
    for (int m = threadIdx.x; m < NPTS; m += blockDim.x) {
      float v = dist[m];
      if (v > bv) { bv = v; bi = m; }
    }
    rval[threadIdx.x] = bv; ridx[threadIdx.x] = bi;
    __syncthreads();
    for (int st = blockDim.x >> 1; st > 0; st >>= 1) {
      if (threadIdx.x < st) {
        float ov = rval[threadIdx.x+st]; int oi = ridx[threadIdx.x+st];
        if (ov > rval[threadIdx.x] || (ov == rval[threadIdx.x] && oi < ridx[threadIdx.x])) {
          rval[threadIdx.x] = ov; ridx[threadIdx.x] = oi;
        }
      }
      __syncthreads();
    }
    if (threadIdx.x == 0) {
      idx[((size_t)b*NPTS + n)*KNBR + kk] = ridx[0];
      dist[ridx[0]] = -3.4e38f;
    }
    __syncthreads();
  }
}

// ---------------- fused dg1-epilogue + dg2 GEMM + k-max (per 8-point tile)
// g1[c, n*20+k] = leaky(bn1(Y1[c,idx[n,k]] + Y2[c,n]));  x1 = max_k g1
// out2 = leaky(bn2(Wdg2 @ g1));                           x2 = max_k out2
__global__ void k_dg_fused(
    const float* __restrict__ Y1, const float* __restrict__ Y2,
    const int* __restrict__ idx,
    const float* __restrict__ g1G, const float* __restrict__ g1B,
    const float* __restrict__ g1M, const float* __restrict__ g1V,
    const float* __restrict__ Wdg2,
    const float* __restrict__ g2G, const float* __restrict__ g2B,
    const float* __restrict__ g2M, const float* __restrict__ g2V,
    float* __restrict__ x1, float* __restrict__ x2)
{
  int b = blockIdx.y, n0 = blockIdx.x * NT;
  extern __shared__ char dg_sm[];
  half_t* g1  = (half_t*)dg_sm;            // [128][CSTR]
  half_t* wsm = g1 + 128*CSTR;             // [128][128]
  float*  o2  = (float*)(wsm + 128*128);   // [128][COLS]

  for (int i = threadIdx.x; i < 128*128; i += blockDim.x) wsm[i] = (half_t)Wdg2[i];

  const float* Y1b = Y1 + (size_t)b*128*NPTS;
  const float* Y2b = Y2 + (size_t)b*128*NPTS;
  for (int i = threadIdx.x; i < 128*NT; i += blockDim.x) {
    int c = i / NT, ln = i % NT, n = n0 + ln;
    float y2 = Y2b[(size_t)c*NPTS + n];
    float sc = g1G[c] * rsqrtf(g1V[c] + EPSF);
    float sh = g1B[c] - g1M[c]*sc;
    const int* ir = idx + ((size_t)b*NPTS + n)*KNBR;
    float mx = -3.4e38f;
    for (int kk = 0; kk < KNBR; ++kk) {
      float v = leakyf((Y1b[(size_t)c*NPTS + ir[kk]] + y2)*sc + sh);
      mx = fmaxf(mx, v);
      g1[c*CSTR + ln*KNBR + kk] = (half_t)v;
    }
    x1[(size_t)b*128*NPTS + (size_t)c*NPTS + n] = mx;
  }
  __syncthreads();

  int wave = threadIdx.x >> 5, lane = threadIdx.x & 31;
  int g = lane >> 4, lm = lane & 15;
  for (int t = wave; t < 80; t += 8) {           // 8 o-tiles x 10 col-tiles
    int o0 = (t / 10) * 16, q0 = (t % 10) * 16;
    v8f acc = {};
    for (int k0 = 0; k0 < 128; k0 += 32) {
      v16h afr, bfr;
      #pragma unroll
      for (int j = 0; j < 8; ++j) {
        int kb = 2*j + 16*g;
        afr[2*j]   = wsm[(o0+lm)*128 + k0 + kb];
        afr[2*j+1] = wsm[(o0+lm)*128 + k0 + kb + 1];
      }
      #pragma unroll
      for (int j = 0; j < 8; ++j) {
        int c = k0 + 16*g + 2*j;
        bfr[2*j]   = g1[(size_t)c*CSTR + q0 + lm];
        bfr[2*j+1] = g1[(size_t)(c+1)*CSTR + q0 + lm];
      }
      acc = __builtin_amdgcn_wmma_f32_16x16x32_f16(false, afr, false, bfr,
                                                   (short)0, acc, false, false);
    }
    #pragma unroll
    for (int r = 0; r < 8; ++r) {
      int o = o0 + r + 8*g;
      float sc = g2G[o] * rsqrtf(g2V[o] + EPSF);
      o2[o*COLS + q0 + lm] = leakyf((acc[r] - g2M[o])*sc + g2B[o]);
    }
  }
  __syncthreads();

  for (int i = threadIdx.x; i < 128*NT; i += blockDim.x) {
    int c = i / NT, ln = i % NT;
    float mx = -3.4e38f;
    for (int kk = 0; kk < KNBR; ++kk) mx = fmaxf(mx, o2[c*COLS + ln*KNBR + kk]);
    x2[(size_t)b*128*NPTS + (size_t)c*NPTS + n0 + ln] = mx;
  }
}

// ---------------- x3[b,c,n] = max_k leaky(bn(Z1[c,idx2[n,k]] + Z2[c,n]))
__global__ void k_x3(const float* __restrict__ Z1, const float* __restrict__ Z2,
                     const int* __restrict__ idx,
                     const float* __restrict__ bnG, const float* __restrict__ bnB,
                     const float* __restrict__ bnM, const float* __restrict__ bnV,
                     float* __restrict__ x3)
{
  int i = blockIdx.x * blockDim.x + threadIdx.x;
  if (i >= BATCH * 256 * NPTS) return;
  int n = i % NPTS, c = (i / NPTS) % 256, b = i / (NPTS * 256);
  const float* Z1b = Z1 + (size_t)b*256*NPTS + (size_t)c*NPTS;
  float z2 = Z2[(size_t)b*256*NPTS + (size_t)c*NPTS + n];
  float sc = bnG[c] * rsqrtf(bnV[c] + EPSF);
  float sh = bnB[c] - bnM[c]*sc;
  const int* ir = idx + ((size_t)b*NPTS + n)*KNBR;
  float mx = -3.4e38f;
  for (int kk = 0; kk < KNBR; ++kk)
    mx = fmaxf(mx, leakyf((Z1b[ir[kk]] + z2)*sc + sh));
  x3[(size_t)b*256*NPTS + (size_t)c*NPTS + n] = mx;
}

// =====================================================================
struct BN4 { const float *g, *b, *m, *v; };

extern "C" void kernel_launch(void* const* d_in, const int* in_sizes, int n_in,
                              void* d_out, int out_size, void* d_ws, size_t ws_size,
                              hipStream_t stream)
{
  (void)n_in; (void)out_size; (void)ws_size;
  auto F = [&](int i){ return (const float*)d_in[i]; };

  const float *x, *tc1_w,*tc1_b,*tc2_w,*tc2_b,*tc3_w,*tc3_b;
  const float *tf1_w,*tf1_b,*tf2_w,*tf2_b,*tf3_w,*tf3_b;
  const float *c1_w,*c2_w,*dg1_w,*dg2_w,*sn1_w,*c3_w;
  BN4 tbn1,tbn2,tbn3,tbn4,tbn5,bn1,bn2,dgbn1,dgbn2,snbn1,bn3;

  if (in_sizes[0] == BATCH*NPTS*3) {
    // dict-insertion order; bn dicts in {g,b,m,v} order
    auto BNI = [&](int i){ BN4 r{F(i),F(i+1),F(i+2),F(i+3)}; return r; };
    x=F(0);
    tc1_w=F(1); tc1_b=F(2);  tbn1=BNI(3);
    tc2_w=F(7); tc2_b=F(8);  tbn2=BNI(9);
    tc3_w=F(13);tc3_b=F(14); tbn3=BNI(15);
    tf1_w=F(19);tf1_b=F(20); tbn4=BNI(21);
    tf2_w=F(25);tf2_b=F(26); tbn5=BNI(27);
    tf3_w=F(31);tf3_b=F(32);
    c1_w=F(33);  bn1=BNI(34);
    c2_w=F(38);  bn2=BNI(39);
    dg1_w=F(43); dgbn1=BNI(44);
    dg2_w=F(48); dgbn2=BNI(49);
    sn1_w=F(53); snbn1=BNI(54);
    c3_w=F(58);  bn3=BNI(59);
  } else {
    // sorted pytree order; bn dicts in {b,g,m,v} order; 'params' before 'x'
    auto BNS = [&](int i){ BN4 r{F(i+1),F(i),F(i+2),F(i+3)}; return r; };
    bn1=BNS(0); bn2=BNS(4); bn3=BNS(8);
    c1_w=F(12); c2_w=F(13); c3_w=F(14);
    dg1_w=F(15); dg2_w=F(16);
    dgbn1=BNS(17); dgbn2=BNS(21);
    sn1_w=F(25); snbn1=BNS(26);
    tbn1=BNS(30); tbn2=BNS(34); tbn3=BNS(38); tbn4=BNS(42); tbn5=BNS(46);
    tc1_b=F(50); tc1_w=F(51); tc2_b=F(52); tc2_w=F(53); tc3_b=F(54); tc3_w=F(55);
    tf1_b=F(56); tf1_w=F(57); tf2_b=F(58); tf2_w=F(59); tf3_b=F(60); tf3_w=F(61);
    x=F(62);
  }

  // ---- workspace carve-up
  char* wsp = (char*)d_ws; size_t off = 0;
  auto allocF = [&](size_t n){ float* p = (float*)(wsp + off); off += ((n*4 + 255)/256)*256; return p; };
  auto allocI = [&](size_t n){ int*   p = (int*)  (wsp + off); off += ((n*4 + 255)/256)*256; return p; };
  const size_t BN_ = (size_t)BATCH*NPTS;
  float* pts  = allocF(BN_*3);
  float* t1   = allocF(BN_*64);
  float* t2   = allocF(BN_*128);
  float* gmax = allocF((size_t)BATCH*1024);
  float* fc1o = allocF((size_t)BATCH*512);
  float* fc2o = allocF((size_t)BATCH*256);
  float* trn  = allocF((size_t)BATCH*9);
  float* h0   = allocF(BN_*3);
  float* h1   = allocF(BN_*64);
  float* h2   = allocF(BN_*64);
  float* Y1   = allocF(BN_*128);
  float* Y2   = allocF(BN_*128);
  float* X1   = allocF(BN_*128);
  float* X2   = allocF(BN_*128);
  float* Z1   = allocF(BN_*256);
  float* Z2   = allocF(BN_*256);
  float* X3   = allocF(BN_*256);
  int*  idx1  = allocI(BN_*KNBR);
  int*  idx2  = allocI(BN_*KNBR);
  float* outp = (float*)d_out;

  const int KNN_SM = NPTS*4 + 64*4 + 256*4 + 256*4;
  const int DG_SM  = 128*CSTR*2 + 128*128*2 + 128*COLS*4;
  hipFuncSetAttribute((const void*)k_dg_fused, hipFuncAttributeMaxDynamicSharedMemorySize, DG_SM);
  hipFuncSetAttribute((const void*)k_knn,      hipFuncAttributeMaxDynamicSharedMemorySize, KNN_SM);

  auto gemm = [&](const float* W, int w_ld, int w_off,
                  const float* i0, size_t s0_, const float* i1, size_t s1_,
                  const float* i2, size_t s2_, int cs1, int cs2,
                  const float* bias, BN4 bn, bool use_bn, int act,
                  float* out, size_t obs, int O, int C){
    dim3 grid(NPTS/32, O/32, BATCH);
    k_gemm_wmma<<<grid, 32, 0, stream>>>(W, w_ld, w_off, i0, s0_, i1, s1_, i2, s2_,
        cs1, cs2, bias,
        use_bn ? bn.g : nullptr, use_bn ? bn.b : nullptr,
        use_bn ? bn.m : nullptr, use_bn ? bn.v : nullptr,
        act, out, obs, O, C, NPTS);
  };
  BN4 nobn{nullptr,nullptr,nullptr,nullptr};

  // 1) transpose input
  k_transpose<<<(BATCH*NPTS+255)/256, 256, 0, stream>>>(x, pts);

  // 2) TNet
  k_conv_small<<<(BATCH*64*NPTS+255)/256, 256, 0, stream>>>(
      tc1_w, tc1_b, tbn1.g,tbn1.b,tbn1.m,tbn1.v, 1, pts, t1, 64);
  gemm(tc2_w, 64, 0, t1, (size_t)64*NPTS, t1, (size_t)64*NPTS, t1, (size_t)64*NPTS,
       64, 64, tc2_b, tbn2, true, 1, t2, (size_t)128*NPTS, 128, 64);
  k_conv_max<<<dim3(1024, BATCH), 256, (128+256)*4, stream>>>(
      tc3_w, tc3_b, tbn3.g, tbn3.b, tbn3.m, tbn3.v, t2, gmax, 1024, 128, NPTS);
  k_fc<<<(BATCH*512+255)/256, 256, 0, stream>>>(tf1_w, tf1_b, tbn4.g,tbn4.b,tbn4.m,tbn4.v, 1, 0, gmax, fc1o, 512, 1024);
  k_fc<<<(BATCH*256+255)/256, 256, 0, stream>>>(tf2_w, tf2_b, tbn5.g,tbn5.b,tbn5.m,tbn5.v, 1, 0, fc1o, fc2o, 256, 512);
  k_fc<<<1, 64, 0, stream>>>(tf3_w, tf3_b, nullptr,nullptr,nullptr,nullptr, 0, 1, fc2o, trn, 9, 256);
  k_apply_trans<<<(BATCH*NPTS+255)/256, 256, 0, stream>>>(pts, trn, h0);

  // 3) stem convs
  k_conv_small<<<(BATCH*64*NPTS+255)/256, 256, 0, stream>>>(
      c1_w, nullptr, bn1.g,bn1.b,bn1.m,bn1.v, 2, h0, h1, 64);
  gemm(c2_w, 64, 0, h1, (size_t)64*NPTS, h1, (size_t)64*NPTS, h1, (size_t)64*NPTS,
       64, 64, nullptr, bn2, true, 2, h2, (size_t)64*NPTS, 64, 64);

  // 4) kNN on feature space (fp32) + dg1 decomposition GEMMs
  k_knn<<<dim3(NPTS, BATCH), 256, KNN_SM, stream>>>(h2, 64, idx1);
  gemm(dg1_w, 128, 0,  h2, (size_t)64*NPTS, h2, (size_t)64*NPTS, h2, (size_t)64*NPTS,
       64, 64, nullptr, nobn, false, 0, Y1, (size_t)128*NPTS, 128, 64);
  gemm(dg1_w, 128, 64, h2, (size_t)64*NPTS, h2, (size_t)64*NPTS, h2, (size_t)64*NPTS,
       64, 64, nullptr, nobn, false, 0, Y2, (size_t)128*NPTS, 128, 64);

  // 5) kNN on original points (for sn1)
  k_knn<<<dim3(NPTS, BATCH), 256, KNN_SM, stream>>>(pts, 3, idx2);

  // 6) fused dg1-activation + dg2 GEMM + k-max -> x1, x2
  k_dg_fused<<<dim3(NPTS/NT, BATCH), 256, DG_SM, stream>>>(
      Y1, Y2, idx1, dgbn1.g, dgbn1.b, dgbn1.m, dgbn1.v,
      dg2_w, dgbn2.g, dgbn2.b, dgbn2.m, dgbn2.v, X1, X2);

  // 7) sn1 decomposition + x3
  gemm(sn1_w, 256, 0,   X2, (size_t)128*NPTS, X2, (size_t)128*NPTS, X2, (size_t)128*NPTS,
       128, 128, nullptr, nobn, false, 0, Z1, (size_t)256*NPTS, 256, 128);
  gemm(sn1_w, 256, 128, X2, (size_t)128*NPTS, X2, (size_t)128*NPTS, X2, (size_t)128*NPTS,
       128, 128, nullptr, nobn, false, 0, Z2, (size_t)256*NPTS, 256, 128);
  k_x3<<<(BATCH*256*NPTS+255)/256, 256, 0, stream>>>(
      Z1, Z2, idx2, snbn1.g, snbn1.b, snbn1.m, snbn1.v, X3);

  // 8) final conv3 over virtual concat [x1|x2|x3]
  gemm(c3_w, 512, 0, X1, (size_t)128*NPTS, X2, (size_t)128*NPTS, X3, (size_t)256*NPTS,
       128, 256, nullptr, bn3, true, 2, outp, (size_t)512*NPTS, 512, 512);
}
